// MultiHeadSelfAttention_41755672052049
// MI455X (gfx1250) — matrix operand order repaired and checked by reading, not compiled, not run on hardware
//
#include <hip/hip_runtime.h>

// ---------------------------------------------------------------------------
// Relative-position multi-head self-attention (Transformer-XL style) for
// MI455X / gfx1250.  All GEMMs run on v_wmma_f32_16x16x32_bf16 (wave32).
// Flash-style streaming attention: scores never touch HBM.  K/V tiles are
// staged into LDS with gfx1250 async global->LDS DMA (double buffered).
// ---------------------------------------------------------------------------

#define B_BATCH 8
#define T_SEQ   1024
#define D_MDL   512
#define NH      8
#define DH      64
#define PPITCH  2048          // padded rows for projected pos encodings
#define SCALE_F 0.125f        // 1/sqrt(64)

typedef __attribute__((ext_vector_type(16))) __bf16 v16bf;
typedef __attribute__((ext_vector_type(8)))  __bf16 v8bf;
typedef __attribute__((ext_vector_type(8)))  float  v8f;
typedef int v4i_vec __attribute__((vector_size(16)));   // b128 payload type

#if defined(__has_builtin)
#define CDNA5_HAS(x) __has_builtin(x)
#else
#define CDNA5_HAS(x) 0
#endif

#if CDNA5_HAS(__builtin_amdgcn_global_load_async_to_lds_b128)
#define HAVE_ASYNC_LDS 1
#else
#define HAVE_ASYNC_LDS 0
#endif

static __device__ __forceinline__ v16bf mk16(v8bf a, v8bf b) {
  v16bf r;
#pragma unroll
  for (int i = 0; i < 8; ++i) { r[i] = a[i]; r[i + 8] = b[i]; }
  return r;
}
// 16 contiguous bf16 (32B) -> one wave-lane WMMA operand fragment
static __device__ __forceinline__ v16bf ld16(const __bf16* p) {
  return mk16(*(const v8bf*)p, *(const v8bf*)(p + 8));
}

static __device__ __forceinline__ v8f wmma_bf16(v16bf a, v16bf b, v8f c) {
  return __builtin_amdgcn_wmma_f32_16x16x32_bf16(false, a, false, b,
                                                 (short)0, c, false, false);
}

// one 16-byte async (or fallback sync) global -> LDS move
static __device__ __forceinline__ void g2l_b128(const __bf16* g, __bf16* l) {
#if HAVE_ASYNC_LDS
  __builtin_amdgcn_global_load_async_to_lds_b128(
      (__attribute__((address_space(1))) v4i_vec*)g,
      (__attribute__((address_space(3))) v4i_vec*)l, 0, 0);
#else
  *(v8bf*)l = *(const v8bf*)g;
#endif
}

template <int N>
static __device__ __forceinline__ void wait_async() {
#if HAVE_ASYNC_LDS
#if CDNA5_HAS(__builtin_amdgcn_s_wait_asynccnt)
  __builtin_amdgcn_s_wait_asynccnt(N);
#else
  asm volatile("s_wait_asynccnt %0" ::"i"(N) : "memory");
#endif
#endif
}

// ---------------------------------------------------------------------------
// fp32 -> bf16 casts / transposes (tiny, bandwidth trivial)
// ---------------------------------------------------------------------------
__global__ void __launch_bounds__(256)
cast_bf16_kernel(const float* __restrict__ in, __bf16* __restrict__ out, int n) {
  int i = blockIdx.x * 256 + threadIdx.x;
  if (i < n) out[i] = (__bf16)in[i];
}

__global__ void __launch_bounds__(256)
pad_pos_kernel(const float* __restrict__ in, __bf16* __restrict__ out) {
  int i = blockIdx.x * 256 + threadIdx.x;            // 2048*512 threads
  float v = (i < (2 * T_SEQ - 1) * D_MDL) ? in[i] : 0.0f;
  out[i] = (__bf16)v;
}

__global__ void __launch_bounds__(256)
transpose_cast_kernel(const float* __restrict__ in, __bf16* __restrict__ out) {
  int i = blockIdx.x * 256 + threadIdx.x;            // 512*512 threads
  int k = i >> 9, nn = i & 511;
  out[(size_t)nn * D_MDL + k] = (__bf16)in[i];       // Wt[n][k] = W[k][n]
}

// ---------------------------------------------------------------------------
// Generic bf16 WMMA GEMM:  C(M x 512) = A(M x 512) @ W(512 x 512) + bias
// Wt is pre-transposed [N][K] so B-operand lanes read 32B contiguous.
// One wave computes a 16x64 strip: 4 v8f accumulators, K-loop of 16x32 steps.
// MODE epilogues scatter into attention-friendly layouts.
//   0: Q -> write q+bias_u and q+bias_v, bf16 [B,H,T,DH]
//   1: K -> bf16 [B,H,T,DH]
//   2: V -> bf16 [B,H,DH,T]   (transposed so PV B-operand is contiguous)
//   3: P -> bf16 [H,PPITCH,DH] (no bias)
//   4: final out -> fp32 [B,T,D] (+bo)
// ---------------------------------------------------------------------------
template <int MODE>
__global__ void __launch_bounds__(128)
gemm_bf16_wmma(const __bf16* __restrict__ A, const __bf16* __restrict__ Wt,
               const float* __restrict__ bias, int M,
               void* __restrict__ out0, void* __restrict__ out1,
               const float* __restrict__ pbu, const float* __restrict__ pbv) {
  const int lane = threadIdx.x & 31;
  const int wave = threadIdx.x >> 5;
  const int m0 = (blockIdx.x * 4 + wave) * 16;
  const int n0 = blockIdx.y * 64;
  if (m0 >= M) return;
  const int n  = lane & 15;
  const int hl = lane >> 4;
  const int kA = hl ? 16 : 0;   // A-operand K base (contiguous, matches B)
  const int kB = hl ? 16 : 0;   // B-operand K base

  const __bf16* arow = A + (size_t)(m0 + n) * D_MDL;
  v8f acc[4];
#pragma unroll
  for (int j = 0; j < 4; ++j) acc[j] = (v8f){0, 0, 0, 0, 0, 0, 0, 0};

  for (int kc = 0; kc < D_MDL; kc += 32) {
    v16bf Aop = mk16(*(const v8bf*)(arow + kc + kA),
                     *(const v8bf*)(arow + kc + kA + 8));
#pragma unroll
    for (int j = 0; j < 4; ++j) {
      const __bf16* wrow = Wt + (size_t)(n0 + j * 16 + n) * D_MDL + kc + kB;
      acc[j] = wmma_bf16(Aop, ld16(wrow), acc[j]);
    }
  }

#pragma unroll
  for (int j = 0; j < 4; ++j) {
    const int nn = n0 + j * 16 + n;
    const float bz = (MODE == 3) ? 0.0f : bias[nn];
#pragma unroll
    for (int r = 0; r < 8; ++r) {
      const int m = m0 + r + hl * 8;          // C/D layout: vgpr r, lane half
      float val = acc[j][r] + bz;
      if (MODE == 0) {
        int h = nn >> 6, d = nn & 63, b = m >> 10, t = m & (T_SEQ - 1);
        size_t idx = (((size_t)b * NH + h) * T_SEQ + t) * DH + d;
        ((__bf16*)out0)[idx] = (__bf16)(val + pbu[h * DH + d]);
        ((__bf16*)out1)[idx] = (__bf16)(val + pbv[h * DH + d]);
      } else if (MODE == 1) {
        int h = nn >> 6, d = nn & 63, b = m >> 10, t = m & (T_SEQ - 1);
        ((__bf16*)out0)[(((size_t)b * NH + h) * T_SEQ + t) * DH + d] = (__bf16)val;
      } else if (MODE == 2) {
        int h = nn >> 6, d = nn & 63, b = m >> 10, t = m & (T_SEQ - 1);
        ((__bf16*)out0)[(((size_t)b * NH + h) * DH + d) * T_SEQ + t] = (__bf16)val;
      } else if (MODE == 3) {
        int h = nn >> 6, d = nn & 63;
        ((__bf16*)out0)[((size_t)h * PPITCH + m) * DH + d] = (__bf16)val;
      } else {
        ((float*)out0)[(size_t)m * D_MDL + nn] = val;
      }
    }
  }
}

// ---------------------------------------------------------------------------
// Fused flash relative-position attention.
// Grid: (T/64, B*H), 128 threads (4 wave32).  Each wave owns a 16-row query
// tile; the workgroup shares K/V s-tiles through LDS, filled by async
// global->LDS DMA with double buffering (overlaps DMA with WMMAs).
// pos term uses identity shifted[t,s] = q_v[t] . p[s + T-1 - t]:
// per 16-col s-tile compute a 16x32 window WMMA of q_v @ p^T, then extract
// the diagonal band with cross-lane shuffles (ds_bpermute).
// ---------------------------------------------------------------------------
__global__ void __launch_bounds__(128)
flash_relattn(const __bf16* __restrict__ qu, const __bf16* __restrict__ qv,
              const __bf16* __restrict__ kb, const __bf16* __restrict__ vtb,
              const __bf16* __restrict__ pb, __bf16* __restrict__ ob) {
  __shared__ __bf16 ldsK[2][32 * DH];        // K tile: 32 rows x 64 (8 KB)
  __shared__ __bf16 ldsV[2][DH * 32];        // V^T tile: 64 rows x 32 (8 KB)
  __shared__ __bf16 lds_probs[4][16 * 32];   // per-wave probs transpose (4 KB)

  const int tid  = threadIdx.x;
  const int lane = tid & 31;
  const int wave = tid >> 5;
  const int bh = blockIdx.y;
  const int b = bh >> 3, h = bh & 7;
  const int t0 = (blockIdx.x * 4 + wave) * 16;
  const int n  = lane & 15;
  const int hl = lane >> 4;
  const int kA = hl ? 16 : 0;
  const int kB = hl ? 16 : 0;

  const size_t bhT = (size_t)bh * T_SEQ;
  const __bf16* kbase = kb + bhT * DH;
  const __bf16* vbase = vtb + (size_t)bh * DH * T_SEQ;
  const __bf16* pbase = pb + (size_t)h * PPITCH * DH;

  // Query operands (A fragments), kept resident: q_u for content, q_v for pos
  v16bf Aqu[2], Aqv[2];
  {
    const __bf16* r0 = qu + (bhT + t0 + n) * DH;
    const __bf16* r1 = qv + (bhT + t0 + n) * DH;
#pragma unroll
    for (int kc = 0; kc < 2; ++kc) {
      int off = kc * 32 + kA;
      Aqu[kc] = mk16(*(const v8bf*)(r0 + off), *(const v8bf*)(r0 + off + 8));
      Aqv[kc] = mk16(*(const v8bf*)(r1 + off), *(const v8bf*)(r1 + off + 8));
    }
  }

  v8f acc[4];
#pragma unroll
  for (int j = 0; j < 4; ++j) acc[j] = (v8f){0, 0, 0, 0, 0, 0, 0, 0};
  float mrow[8], lrow[8];
#pragma unroll
  for (int r = 0; r < 8; ++r) { mrow[r] = -1e30f; lrow[r] = 0.0f; }

  // --- cooperative staging: 128 threads x (2 K + 2 V) 16B chunks per tile ---
  auto stage = [&](int buf, int s0) {
#pragma unroll
    for (int c2 = 0; c2 < 2; ++c2) {
      int c = tid * 2 + c2;                       // 0..255
      int kr = c >> 3, kcol = (c & 7) * 8;        // K: 32 rows x 128B
      g2l_b128(kbase + (size_t)(s0 + kr) * DH + kcol,
               &ldsK[buf][kr * DH + kcol]);
      int vr = c >> 2, vcol = (c & 3) * 8;        // V^T: 64 rows x 64B
      g2l_b128(vbase + (size_t)vr * T_SEQ + s0 + vcol,
               &ldsV[buf][vr * 32 + vcol]);
    }
  };

  stage(0, 0);

  const int NIT = T_SEQ / 32;
  for (int it = 0; it < NIT; ++it) {
    const int s0 = it * 32;
    const int cur = it & 1;
    if (it + 1 < NIT) {
      stage(cur ^ 1, s0 + 32);   // DMA next tile while computing this one
      wait_async<4>();           // this wave's loads for buf[cur] arrived
    } else {
      wait_async<0>();
    }
    __syncthreads();             // all waves' buf[cur] DMA arrived

    const __bf16* lK = ldsK[cur];
    const __bf16* lV = ldsV[cur];

    float sc[2][8];
#pragma unroll
    for (int st = 0; st < 2; ++st) {
      const int sT = s0 + st * 16;
      // ---- content scores: (q+u) @ k^T, 16x16 tile, K=64 in two steps ----
      v8f c = (v8f){0, 0, 0, 0, 0, 0, 0, 0};
      const __bf16* krow = lK + (st * 16 + n) * DH + kB;
#pragma unroll
      for (int kc = 0; kc < 2; ++kc)
        c = wmma_bf16(Aqu[kc], ld16(krow + kc * 32), c);

      // ---- pos scores: 16x32 window of (q+v) @ p^T, rows u0 .. u0+31 ----
      const int u0 = sT + (T_SEQ - 1) - t0 - 15;   // always in [0, 2016]
      v8f w0 = (v8f){0, 0, 0, 0, 0, 0, 0, 0};
      v8f w1 = (v8f){0, 0, 0, 0, 0, 0, 0, 0};
      const __bf16* prow0 = pbase + (size_t)(u0 + n) * DH + kB;
      const __bf16* prow1 = pbase + (size_t)(u0 + 16 + n) * DH + kB;
#pragma unroll
      for (int kc = 0; kc < 2; ++kc) {
        w0 = wmma_bf16(Aqv[kc], ld16(prow0 + kc * 32), w0);
        w1 = wmma_bf16(Aqv[kc], ld16(prow1 + kc * 32), w1);
      }
      // ---- diagonal band extraction: row i needs window col n + 15 - i ----
#pragma unroll
      for (int r = 0; r < 8; ++r) {
        int i = r + hl * 8;
        int cidx = n + 15 - i;                       // 0..30
        int src = (cidx & 15) | (lane & 16);         // stay in lane half
        float v0 = __shfl(w0[r], src, 32);
        float v1 = __shfl(w1[r], src, 32);
        float pv = (cidx < 16) ? v0 : v1;
        sc[st][r] = (c[r] + pv) * SCALE_F;
      }
    }

    // ---- online softmax over the 32 new columns ----
    float fac[8];
#pragma unroll
    for (int r = 0; r < 8; ++r) {
      float t = fmaxf(sc[0][r], sc[1][r]);
#pragma unroll
      for (int off = 8; off >= 1; off >>= 1) t = fmaxf(t, __shfl_xor(t, off, 16));
      float mnew = fmaxf(mrow[r], t);
      fac[r] = __expf(mrow[r] - mnew);
      mrow[r] = mnew;
      sc[0][r] = __expf(sc[0][r] - mnew);
      sc[1][r] = __expf(sc[1][r] - mnew);
      float rs = sc[0][r] + sc[1][r];
#pragma unroll
      for (int off = 8; off >= 1; off >>= 1) rs += __shfl_xor(rs, off, 16);
      lrow[r] = lrow[r] * fac[r] + rs;
    }
#pragma unroll
    for (int j = 0; j < 4; ++j)
#pragma unroll
      for (int r = 0; r < 8; ++r) acc[j][r] *= fac[r];

    // ---- probs (C-layout) -> LDS -> A-operand layout for PV WMMA ----
    __bf16* lp = &lds_probs[wave][0];
#pragma unroll
    for (int r = 0; r < 8; ++r) {
      int row = r + hl * 8;
      lp[row * 32 + n]      = (__bf16)sc[0][r];
      lp[row * 32 + 16 + n] = (__bf16)sc[1][r];
    }
    asm volatile("s_wait_dscnt 0" ::: "memory");
    v16bf Ap = mk16(*(const v8bf*)(lp + n * 32 + kA),
                    *(const v8bf*)(lp + n * 32 + kA + 8));

    // ---- acc += probs(16x32) @ V(32x64); V^T rows read from LDS ----
#pragma unroll
    for (int j = 0; j < 4; ++j) {
      const __bf16* vrow = lV + (j * 16 + n) * 32 + kB;
      acc[j] = wmma_bf16(Ap, ld16(vrow), acc[j]);
    }

    __syncthreads();             // everyone done reading buf[cur]
  }

  // ---- normalize and scatter to [B,T,D] bf16 for the output projection ----
#pragma unroll
  for (int r = 0; r < 8; ++r) {
    float inv = 1.0f / lrow[r];
    int tt = t0 + r + hl * 8;
    __bf16* orow = ob + ((size_t)b * T_SEQ + tt) * D_MDL + h * DH;
#pragma unroll
    for (int j = 0; j < 4; ++j) orow[j * 16 + n] = (__bf16)(acc[j][r] * inv);
  }
}

// ---------------------------------------------------------------------------
extern "C" void kernel_launch(void* const* d_in, const int* in_sizes, int n_in,
                              void* d_out, int out_size, void* d_ws, size_t ws_size,
                              hipStream_t stream) {
  (void)in_sizes; (void)n_in; (void)out_size; (void)ws_size;
  const float* x    = (const float*)d_in[0];
  const float* pos  = (const float*)d_in[1];
  const float* Wq   = (const float*)d_in[2];
  const float* bq   = (const float*)d_in[3];
  const float* Wk   = (const float*)d_in[4];
  const float* bk   = (const float*)d_in[5];
  const float* Wv   = (const float*)d_in[6];
  const float* bv   = (const float*)d_in[7];
  const float* Wpos = (const float*)d_in[8];
  const float* pbu  = (const float*)d_in[9];
  const float* pbv  = (const float*)d_in[10];
  const float* Wo   = (const float*)d_in[11];
  const float* bo   = (const float*)d_in[12];

  char* ws = (char*)d_ws;
  size_t off = 0;
  auto alloc = [&](size_t bytes) -> void* {
    void* p = ws + off;
    off += (bytes + 255) & ~(size_t)255;
    return p;
  };
  const size_t NTOK = (size_t)B_BATCH * T_SEQ;            // 8192
  __bf16* xb  = (__bf16*)alloc(NTOK * D_MDL * 2);         // x in bf16
  __bf16* pe  = (__bf16*)alloc((size_t)PPITCH * D_MDL * 2);
  __bf16* Wqt = (__bf16*)alloc((size_t)D_MDL * D_MDL * 2);
  __bf16* Wkt = (__bf16*)alloc((size_t)D_MDL * D_MDL * 2);
  __bf16* Wvt = (__bf16*)alloc((size_t)D_MDL * D_MDL * 2);
  __bf16* Wpt = (__bf16*)alloc((size_t)D_MDL * D_MDL * 2);
  __bf16* Wot = (__bf16*)alloc((size_t)D_MDL * D_MDL * 2);
  __bf16* qub = (__bf16*)alloc(NTOK * D_MDL * 2);         // [B,H,T,DH]
  __bf16* qvb = (__bf16*)alloc(NTOK * D_MDL * 2);
  __bf16* kbf = (__bf16*)alloc(NTOK * D_MDL * 2);         // [B,H,T,DH]
  __bf16* vtb = (__bf16*)alloc(NTOK * D_MDL * 2);         // [B,H,DH,T]
  __bf16* ppj = (__bf16*)alloc((size_t)NH * PPITCH * DH * 2); // [H,2048,DH]
  __bf16* obf = (__bf16*)alloc(NTOK * D_MDL * 2);         // [B,T,D]

  const int nx = (int)(NTOK * D_MDL);
  cast_bf16_kernel<<<nx / 256, 256, 0, stream>>>(x, xb, nx);
  pad_pos_kernel<<<PPITCH * D_MDL / 256, 256, 0, stream>>>(pos, pe);
  transpose_cast_kernel<<<D_MDL * D_MDL / 256, 256, 0, stream>>>(Wq, Wqt);
  transpose_cast_kernel<<<D_MDL * D_MDL / 256, 256, 0, stream>>>(Wk, Wkt);
  transpose_cast_kernel<<<D_MDL * D_MDL / 256, 256, 0, stream>>>(Wv, Wvt);
  transpose_cast_kernel<<<D_MDL * D_MDL / 256, 256, 0, stream>>>(Wpos, Wpt);
  transpose_cast_kernel<<<D_MDL * D_MDL / 256, 256, 0, stream>>>(Wo, Wot);

  dim3 ggrid(128, 8), gblk(128);
  gemm_bf16_wmma<0><<<ggrid, gblk, 0, stream>>>(xb, Wqt, bq, (int)NTOK,
                                                qub, qvb, pbu, pbv);
  gemm_bf16_wmma<1><<<ggrid, gblk, 0, stream>>>(xb, Wkt, bk, (int)NTOK,
                                                kbf, nullptr, nullptr, nullptr);
  gemm_bf16_wmma<2><<<ggrid, gblk, 0, stream>>>(xb, Wvt, bv, (int)NTOK,
                                                vtb, nullptr, nullptr, nullptr);
  gemm_bf16_wmma<3><<<dim3(PPITCH / 64, 8), gblk, 0, stream>>>(
      pe, Wpt, nullptr, PPITCH, ppj, nullptr, nullptr, nullptr);

  flash_relattn<<<dim3(T_SEQ / 64, B_BATCH * NH), gblk, 0, stream>>>(
      qub, qvb, kbf, vtb, ppj, obf);

  gemm_bf16_wmma<4><<<ggrid, gblk, 0, stream>>>(obf, Wot, bo, (int)NTOK,
                                                d_out, nullptr, nullptr, nullptr);
}
